// FE_block_3d_83958020702877
// MI455X (gfx1250) — matrix operand order repaired and checked by reading, not compiled, not run on hardware
//
#include <hip/hip_runtime.h>

// CDNA5 / gfx1250 implicit-GEMM 3D-conv (4-branch light-field FE block) using
// V_WMMA_F32_16X16X32_F16.  K=32 per WMMA == C_in; 27 taps fully unrolled;
// branch geometry is a template parameter so all tap offsets are immediates.

typedef __attribute__((ext_vector_type(16))) _Float16 v16h;
typedef __attribute__((ext_vector_type(8)))  float    v8f;

#define B_   8
#define C_   32
#define U_   5
#define V_   5
#define H_   64
#define W_   64
#define OC_  8
#define TAPS 27

#define SC (U_*V_*H_*W_)   /* 102400  channel stride   */
#define SB (C_*SC)         /* 3276800 batch stride     */
#define SU (V_*H_*W_)      /* 20480                    */
#define SV (H_*W_)         /* 4096                    */
#define SH (W_)            /* 64                      */

#define TILES_PER_BRANCH (B_*U_*V_*H_*(W_/16))               /* 51200 */
#define WAVES_PER_BLOCK  8
#define BLOCKS_PER_BRANCH (TILES_PER_BRANCH/WAVES_PER_BLOCK) /* 6400  */

template <int BR>
__global__ __launch_bounds__(256) void fe4d_wmma_kernel(
    const float* __restrict__ x,
    const float* __restrict__ wbr,
    const float* __restrict__ bbr,
    const float* __restrict__ abr,
    float* __restrict__ out)
{
  // Weights pre-arranged in WMMA B-matrix per-lane layout:
  // wBsh[t][lane][j] = W[oc = lane&15][c = (lane<16?0:8)+(j<8?j:j+8)][tap t]  (0 for oc>=8)
  __shared__ __attribute__((aligned(32))) _Float16 wBsh[TAPS][32][16];
  __shared__ float trans[WAVES_PER_BLOCK][16][17];   // padded, per-wave D tile

  const float alpha = abr[0];

  // ---- stage branch weights into LDS (f32 -> f16) ----
  for (int e = threadIdx.x; e < TAPS*32*16; e += 256) {
    int t    = e >> 9;           // tap
    int rem  = e & 511;
    int ln   = rem >> 4;         // lane slot
    int j    = rem & 15;         // K index within half
    int c    = ((ln < 16) ? 0 : 8) + ((j < 8) ? j : j + 8);
    int oc   = ln & 15;
    float wv = (oc < OC_) ? wbr[(oc*C_ + c)*TAPS + t] : 0.0f;
    wBsh[t][ln][j] = (_Float16)wv;
  }
  __syncthreads();

  const int wave = threadIdx.x >> 5;
  const int lane = threadIdx.x & 31;

  // ---- tile coordinates: 16 consecutive W positions ----
  int tile = blockIdx.x * WAVES_PER_BLOCK + wave;      // < 51200 exactly
  int tw = tile & 3; int r = tile >> 2;
  int h  = r & 63;   r >>= 6;
  int v  = r % 5;    r /= 5;
  int u  = r % 5;    r /= 5;
  int b  = r;
  const int w0 = tw * 16;

  const int m    = lane & 15;               // A-matrix row = position in tile
  const int coff = (lane >= 16) ? 8 : 0;    // K-half select per ISA A layout
  const int wpos = w0 + m;

  const float* basep =
      x + (size_t)b*SB + (size_t)u*SU + (size_t)v*SV + (size_t)h*SH + wpos;

  v8f acc = {};

  #pragma unroll
  for (int t = 0; t < TAPS; ++t) {
    const int o0 = t / 9 - 1;          // compile-time after unroll
    const int o1 = (t / 3) % 3 - 1;
    const int o2 = t % 3 - 1;
    const int du = (BR == 3) ? 0 : o0;
    const int dv = (BR <= 1) ? o1 : ((BR == 2) ? 0 : o0);
    const int dh = (BR == 0) ? o2 : ((BR == 1) ? 0 : o1);
    const int dw = (BR == 0) ? 0 : o2;

    // zero-padding: merge uniform (u/v/h) and per-lane (w) checks so the
    // WMMA itself is never under an EXEC mask.
    bool valid = ((unsigned)(u + du)    < (unsigned)U_) &
                 ((unsigned)(v + dv)    < (unsigned)V_) &
                 ((unsigned)(h + dh)    < (unsigned)H_) &
                 ((unsigned)(wpos + dw) < (unsigned)W_);

    v16h a = {};
    if (valid) {
      const float* src = basep + (du*SU + dv*SV + dh*SH + dw);
      #pragma unroll
      for (int j = 0; j < 16; ++j) {
        int c = coff + ((j < 8) ? j : j + 8);   // ISA 16-bit A 16x32 layout
        a[j] = (_Float16)src[(size_t)c * SC];   // coalesced across lanes (W)
      }
    }
    v16h bmat = *reinterpret_cast<const v16h*>(&wBsh[t][lane][0]);
    acc = __builtin_amdgcn_wmma_f32_16x16x32_f16(
        false, a, false, bmat, (short)0, acc, false, false);
  }

  // ---- D tile -> LDS transpose for coalesced float4 stores ----
  const int n     = lane & 15;            // column = out channel
  const int mbase = (lane >> 4) * 8;      // rows per ISA C/D layout
  #pragma unroll
  for (int rr = 0; rr < 8; ++rr)
    trans[wave][mbase + rr][n] = acc[rr];
  __syncthreads();

  const int oc  = lane >> 2;              // 0..7
  const int seg = lane & 3;               // 0..3 (4 positions each)
  const float bias = bbr[oc];
  float t0[4];
  #pragma unroll
  for (int i2 = 0; i2 < 4; ++i2) {
    float val = trans[wave][seg*4 + i2][oc] + bias;
    t0[i2] = (val >= 0.0f) ? val : alpha * val;   // PReLU
  }
  float4 o4; o4.x = t0[0]; o4.y = t0[1]; o4.z = t0[2]; o4.w = t0[3];
  size_t oaddr = (size_t)b*SB + (size_t)(BR*OC_ + oc)*SC
               + (size_t)u*SU + (size_t)v*SV + (size_t)h*SH + w0 + seg*4;
  *reinterpret_cast<float4*>(out + oaddr) = o4;
}

extern "C" void kernel_launch(void* const* d_in, const int* in_sizes, int n_in,
                              void* d_out, int out_size, void* d_ws, size_t ws_size,
                              hipStream_t stream) {
  (void)in_sizes; (void)n_in; (void)out_size; (void)d_ws; (void)ws_size;
  const float* x = (const float*)d_in[0];
  float* out = (float*)d_out;
  dim3 grid(BLOCKS_PER_BRANCH, 1, 1);
  dim3 blk(256, 1, 1);
  // input order: x, then (w, b, a) for uvx, uvy, uxy, vxy
  fe4d_wmma_kernel<0><<<grid, blk, 0, stream>>>(
      x, (const float*)d_in[1],  (const float*)d_in[2],  (const float*)d_in[3],  out);
  fe4d_wmma_kernel<1><<<grid, blk, 0, stream>>>(
      x, (const float*)d_in[4],  (const float*)d_in[5],  (const float*)d_in[6],  out);
  fe4d_wmma_kernel<2><<<grid, blk, 0, stream>>>(
      x, (const float*)d_in[7],  (const float*)d_in[8],  (const float*)d_in[9],  out);
  fe4d_wmma_kernel<3><<<grid, blk, 0, stream>>>(
      x, (const float*)d_in[10], (const float*)d_in[11], (const float*)d_in[12], out);
}
